// DiffKMeansMultiClassv2_45861660787183
// MI455X (gfx1250) — matrix operand order repaired and checked by reading, not compiled, not run on hardware
//
#include <hip/hip_runtime.h>
#include <hip/hip_bf16.h>

typedef __attribute__((ext_vector_type(16))) _Float16 v16h;
typedef __attribute__((ext_vector_type(8)))  _Float16 v8h;
typedef __attribute__((ext_vector_type(8)))  float    v8f;

#define DD 256          // feature dim (reference: D=256)
#define KK 64           // clusters per class (reference: K=64)
#define NEG_INF_F (-1.0e12f)
#define DIST_SCALE 6.25f        // 100 / sqrt(256)

// ---------------- k0: init scratch ----------------
__global__ void k0_init(int* __restrict__ cnt, int* __restrict__ cur,
                        float* __restrict__ seg, int* __restrict__ cidx,
                        int C, int idx_len) {
    int t = blockIdx.x * blockDim.x + threadIdx.x;
    if (t < idx_len) cidx[t] = -1;
    if (t < C) { cnt[t] = 0; cur[t] = 0; seg[t] = 0.0f; }
}

// ---------------- k1: normalize -> f16, row norms, label counts ----------------
// One wave per row; lane handles 8 contiguous elements. No LDS, no barriers.
__global__ void __launch_bounds__(256)
k1_norm(const float* __restrict__ data, const float* __restrict__ med,
        const float* __restrict__ stdv, const int* __restrict__ labels,
        _Float16* __restrict__ xn, float* __restrict__ x2,
        int* __restrict__ cnt, int N) {
    int wid  = threadIdx.x >> 5;
    int lane = threadIdx.x & 31;
    int n = blockIdx.x * 8 + wid;
    if (n >= N) return;

    const float4* dr = (const float4*)(data + (size_t)n * DD) + lane * 2;
    const float4* mr = (const float4*)med  + lane * 2;
    const float4* sr = (const float4*)stdv + lane * 2;
    float4 d0 = dr[0], d1 = dr[1];
    float4 m0 = mr[0], m1 = mr[1];
    float4 s0 = sr[0], s1 = sr[1];

    float q[8];
    q[0] = (d0.x - m0.x) / s0.x;  q[1] = (d0.y - m0.y) / s0.y;
    q[2] = (d0.z - m0.z) / s0.z;  q[3] = (d0.w - m0.w) / s0.w;
    q[4] = (d1.x - m1.x) / s1.x;  q[5] = (d1.y - m1.y) / s1.y;
    q[6] = (d1.z - m1.z) / s1.z;  q[7] = (d1.w - m1.w) / s1.w;

    v8h h;
    float ss = 0.0f;
    #pragma unroll
    for (int e = 0; e < 8; e++) { h[e] = (_Float16)q[e]; ss += q[e] * q[e]; }
    *((v8h*)(xn + (size_t)n * DD) + lane) = h;

    #pragma unroll
    for (int off = 16; off > 0; off >>= 1) ss += __shfl_xor(ss, off, 32);
    if (lane == 0) {
        x2[n] = ss;
        atomicAdd(&cnt[labels[n]], 1);
    }
}

// ---------------- k2: mu -> f16, ||mu||^2, tau, log(tau), valid mask ----------------
__global__ void __launch_bounds__(256)
k2_mu(const float* __restrict__ mu, _Float16* __restrict__ muf,
      float* __restrict__ m2, const float* __restrict__ exp_temp,
      float* __restrict__ tau, float* __restrict__ ltau,
      const float* __restrict__ ra, const float* __restrict__ rb,
      float* __restrict__ valid, int CK) {
    int wid  = threadIdx.x >> 5;
    int lane = threadIdx.x & 31;
    int ck = blockIdx.x * 8 + wid;
    if (ck >= CK) return;

    const float4* dr = (const float4*)(mu + (size_t)ck * DD) + lane * 2;
    float4 d0 = dr[0], d1 = dr[1];
    float q[8] = {d0.x, d0.y, d0.z, d0.w, d1.x, d1.y, d1.z, d1.w};

    v8h h;
    float ss = 0.0f;
    #pragma unroll
    for (int e = 0; e < 8; e++) { h[e] = (_Float16)q[e]; ss += q[e] * q[e]; }
    *((v8h*)(muf + (size_t)ck * DD) + lane) = h;

    #pragma unroll
    for (int off = 16; off > 0; off >>= 1) ss += __shfl_xor(ss, off, 32);
    if (lane == 0) {
        m2[ck] = ss;
        float e = exp_temp[ck];
        float t = 100.0f / (1.0f + __expf(-e * 0.5f)) + 0.01f;  // sigmoid(e/2)*100 + 1/100
        tau[ck]  = t;
        ltau[ck] = __logf(t);
        int c = ck / KK;
        valid[ck] = (ra[ck] > rb[c] * (0.5f / (float)KK)) ? 1.0f : 0.0f;
    }
}

// ---------------- k3a: tiny scan of padded (x16) class counts ----------------
__global__ void k3_scan(const int* __restrict__ cnt, int* __restrict__ poff,
                        int* __restrict__ tstart, int C) {
    if (blockIdx.x == 0 && threadIdx.x == 0) {
        int acc = 0;
        for (int c = 0; c < C; c++) {
            poff[c]   = acc;
            tstart[c] = acc >> 4;
            acc += (cnt[c] + 15) & ~15;   // pad each bucket to a 16-row tile
        }
        poff[C]   = acc;
        tstart[C] = acc >> 4;
    }
}

// ---------------- k3b: scatter sample indices into class buckets ----------------
__global__ void k3_scatter(const int* __restrict__ labels, const int* __restrict__ poff,
                           int* __restrict__ cur, int* __restrict__ cidx, int N) {
    int n = blockIdx.x * blockDim.x + threadIdx.x;
    if (n >= N) return;
    int c = labels[n];
    int p = atomicAdd(&cur[c], 1);
    cidx[poff[c] + p] = n;
}

// ---------------- k4: 16-sample x 64-centroid WMMA tile + fused epilogue ----------------
__global__ void __launch_bounds__(32)
k4_main(const _Float16* __restrict__ xn, const float* __restrict__ x2,
        const _Float16* __restrict__ muf, const float* __restrict__ m2,
        const float* __restrict__ tau, const float* __restrict__ ltau,
        const float* __restrict__ valid, const int* __restrict__ cidx,
        const int* __restrict__ tstart, float* __restrict__ seg, int C) {
    __shared__ float simL[16 * KK];   // [row][centroid]
    __shared__ float x2L[16];

    int t = blockIdx.x;
    if (t >= tstart[C]) return;       // padding blocks retire
    // locate class of this tile (C is tiny)
    int c = C - 1;
    for (int i = 0; i < C; i++) {
        if (t < tstart[i + 1]) { c = i; break; }
    }

    int lane = threadIdx.x;
    int hs   = lane >> 4;     // half-wave select
    int mrow = lane & 15;     // this lane's A-row / B-column index
    int ridx = cidx[t * 16 + mrow];   // gathered sample index (-1 = pad row)

    if (lane < 16) x2L[lane] = (ridx >= 0) ? x2[ridx] : 0.0f;
    __syncthreads();

    // ---- A fragments: 16x32 f16 per k-step, ISA lane layout ----
    // lane holds row M=lane&15; VGPR j<4: K = 8*hs + 2j (+1); VGPR j>=4: K = 16 + 8*hs + 2(j-4)
    union { v16h h; unsigned int u[8]; } a[8];
    if (ridx >= 0) {
        const unsigned int* xr = (const unsigned int*)(xn + (size_t)ridx * DD);
        #pragma unroll
        for (int kk = 0; kk < 8; kk++) {
            #pragma unroll
            for (int j = 0; j < 4; j++) {
                a[kk].u[j]     = xr[kk * 16 + 4 * hs + j];
                a[kk].u[4 + j] = xr[kk * 16 + 8 + 4 * hs + j];
            }
        }
    } else {
        #pragma unroll
        for (int kk = 0; kk < 8; kk++)
            #pragma unroll
            for (int j = 0; j < 8; j++) a[kk].u[j] = 0u;
    }

    const _Float16* mubase = muf + (size_t)c * KK * DD;

    #pragma unroll
    for (int s = 0; s < 4; s++) {                 // 4 N-subtiles of 16 centroids
        int cent = s * 16 + mrow;
        const _Float16* mr = mubase + (size_t)cent * DD;
        v8f acc = {};
        #pragma unroll
        for (int kk = 0; kk < 8; kk++) {          // K = 256 in 8 steps of 32
            // B 32x16: lane holds column N=lane&15, K = 8*hs + t (t<8), 16 + 8*hs + (t-8) (t>=8)
            v8h b0 = *(const v8h*)(mr + kk * 32 + hs * 8);
            v8h b1 = *(const v8h*)(mr + kk * 32 + 16 + hs * 8);
            v16h b = __builtin_shufflevector(b0, b1, 0, 1, 2, 3, 4, 5, 6, 7, 8, 9, 10, 11, 12, 13, 14, 15);
            acc = __builtin_amdgcn_wmma_f32_16x16x32_f16(
                false, a[kk].h, false, b, (short)0, acc, false, false);
        }
        float m2v = m2[c * KK + cent];
        float vv  = valid[c * KK + cent];
        #pragma unroll
        for (int i = 0; i < 8; i++) {             // C frag: lane row M = i + 8*hs
            int m = i + 8 * hs;
            float d2   = x2L[m] + m2v - 2.0f * acc[i];
            float dist = -sqrtf(fmaxf(d2, 0.0f)) * DIST_SCALE;
            simL[m * KK + cent] = (vv > 0.0f) ? dist : NEG_INF_F;
        }
    }
    __syncthreads();

    // ---- fused softmax(0.5*sim) + logpdf weighting ----
    // All 32 lanes active: lane covers row (lane&15), k-range [hs*32, hs*32+32).
    const float* rowsim = simL + mrow * KK + hs * 32;
    const float* tc = tau  + c * KK + hs * 32;
    const float* lc = ltau + c * KK + hs * 32;

    float lmax = -3.4e38f;
    #pragma unroll 4
    for (int k = 0; k < 32; k++) lmax = fmaxf(lmax, 0.5f * rowsim[k]);
    lmax = fmaxf(lmax, __shfl_xor(lmax, 16, 32));   // full-row max in both halves

    float Z = 0.0f, accum = 0.0f;
    #pragma unroll 4
    for (int k = 0; k < 32; k++) {
        float sv = rowsim[k];
        float e  = __expf(0.5f * sv - lmax);
        Z     += e;
        accum += (sv / tc[k] - lc[k]) * e;          // logpdf * unnormalized r
    }
    Z     += __shfl_xor(Z, 16, 32);
    accum += __shfl_xor(accum, 16, 32);

    float per = (ridx >= 0) ? -(accum / Z) : 0.0f;  // pad rows contribute 0
    // butterfly-reduce the 16 rows (rows duplicated across halves, xor<16 stays in half)
    per += __shfl_xor(per, 1, 32);
    per += __shfl_xor(per, 2, 32);
    per += __shfl_xor(per, 4, 32);
    per += __shfl_xor(per, 8, 32);
    if (lane == 0) atomicAdd(&seg[c], per);         // one atomic per block
}

// ---------------- k5: final scalar loss ----------------
__global__ void k5_final(const float* __restrict__ seg, const int* __restrict__ cnt,
                         float* __restrict__ out, int C) {
    if (blockIdx.x == 0 && threadIdx.x == 0) {
        float L = 0.0f;
        for (int c = 0; c < C; c++) {
            int n = cnt[c];
            if (n > 0) L += seg[c] / (float)n;
        }
        out[0] = L;
    }
}

extern "C" void kernel_launch(void* const* d_in, const int* in_sizes, int n_in,
                              void* d_out, int out_size, void* d_ws, size_t ws_size,
                              hipStream_t stream) {
    const float* data   = (const float*)d_in[0];
    const int*   labels = (const int*)d_in[1];
    const float* mu     = (const float*)d_in[2];
    const float* expt   = (const float*)d_in[3];
    const float* med    = (const float*)d_in[4];
    const float* stdv   = (const float*)d_in[5];
    const float* ra     = (const float*)d_in[6];
    const float* rb     = (const float*)d_in[7];

    int D = in_sizes[4];          // 256
    int N = in_sizes[0] / D;      // 131072
    int C = in_sizes[7];          // 20
    (void)n_in; (void)ws_size; (void)out_size;

    // carve workspace (256B aligned blocks)
    char* ws = (char*)d_ws;
    size_t off = 0;
    auto carve = [&](size_t bytes) -> char* {
        char* p = ws + off;
        off += (bytes + 255) & ~(size_t)255;
        return p;
    };
    _Float16* xn   = (_Float16*)carve((size_t)N * DD * sizeof(_Float16));
    float*    x2   = (float*)   carve((size_t)N * sizeof(float));
    _Float16* mf   = (_Float16*)carve((size_t)C * KK * DD * sizeof(_Float16));
    float*    m2   = (float*)   carve((size_t)C * KK * sizeof(float));
    float*    tau  = (float*)   carve((size_t)C * KK * sizeof(float));
    float*    ltau = (float*)   carve((size_t)C * KK * sizeof(float));
    float*    vld  = (float*)   carve((size_t)C * KK * sizeof(float));
    int*      cnt  = (int*)     carve((size_t)C * sizeof(int));
    int*      cur  = (int*)     carve((size_t)C * sizeof(int));
    int*      poff = (int*)     carve((size_t)(C + 1) * sizeof(int));
    int*      tst  = (int*)     carve((size_t)(C + 1) * sizeof(int));
    float*    seg  = (float*)   carve((size_t)C * sizeof(float));
    int idx_len = N + 16 * C;
    int* cidx = (int*)carve((size_t)idx_len * sizeof(int));

    int CK = C * KK;
    k0_init   <<<(idx_len + 255) / 256, 256, 0, stream>>>(cnt, cur, seg, cidx, C, idx_len);
    k1_norm   <<<(N + 7) / 8, 256, 0, stream>>>(data, med, stdv, labels, xn, x2, cnt, N);
    k2_mu     <<<(CK + 7) / 8, 256, 0, stream>>>(mu, mf, m2, expt, tau, ltau, ra, rb, vld, CK);
    k3_scan   <<<1, 1, 0, stream>>>(cnt, poff, tst, C);
    k3_scatter<<<(N + 255) / 256, 256, 0, stream>>>(labels, poff, cur, cidx, N);

    int maxTiles = (N + 15) / 16 + C;   // static grid; padding blocks self-retire
    k4_main   <<<maxTiles, 32, 0, stream>>>(xn, x2, mf, m2, tau, ltau, vld, cidx, tst, seg, C);
    k5_final  <<<1, 32, 0, stream>>>(seg, cnt, (float*)d_out, C);
}
